// RNNFeaturizer_335007449404
// MI455X (gfx1250) — matrix operand order repaired and checked by reading, not compiled, not run on hardware
//
#include <hip/hip_runtime.h>
#include <math.h>

// ---------------------------------------------------------------------------
// LSTM featurizer for MI455X (gfx1250, wave32, WMMA).
//   T=256, B=64, NINP=64, NHID=4096, 4H=16384.
// Strategy:
//  * Convert w_hh/w_ih/emb to bf16 in workspace (w_hh bf16 = 128MB -> L2 resident).
//  * Persistent kernel: 256 WGs x 256 threads; WG owns 16 hidden units (all 4 gates).
//  * Per step: fused input projection (K=64) + recurrent GEMM (K=4096) via
//    v_wmma_f32_16x16x32_bf16; gates -> LDS; elementwise LSTM in f32 registers.
//  * h double-buffered bf16 in workspace; grid-wide barrier per step.
// ---------------------------------------------------------------------------

#define T_STEPS 256
#define BATCH   64
#define NINP    64
#define NHID    4096
#define GATES   (4 * NHID)

// workspace layout (bytes)
#define WHH_OFF 0u                      // 16384*4096 bf16 = 134217728
#define WIH_OFF 134217728u              // 16384*64  bf16 = 2097152
#define EMB_OFF 136314880u              // 256*64*64 bf16 = 2097152
#define HBUF_OFF 138412032u             // 2 * 64*4096 bf16 = 1048576
#define CNT_OFF 139460608u              // 1 x u32

typedef __attribute__((ext_vector_type(16))) __bf16 v16bf;
typedef __attribute__((ext_vector_type(8)))  float  v8f;

union FragU {
    v16bf v;
    uint4 u[2];
};

static __device__ __forceinline__ unsigned short f2bf(float f) {
    unsigned int u = __float_as_uint(f);
    u += 0x7FFFu + ((u >> 16) & 1u);    // round-to-nearest-even
    return (unsigned short)(u >> 16);
}

static __device__ __forceinline__ float sigm(float x) {
    return 1.0f / (1.0f + __expf(-x));
}

// One K=32 step: loads A-fragment (rows m0..m0+15) and two B-fragments
// (16 columns each at col0/col1), issues two bf16 WMMAs.
static __device__ __forceinline__ void kstep(
    const unsigned short* __restrict__ A, int lda,
    const unsigned short* __restrict__ B, int ldb,
    int col0, int col1, int kc, int m0, int nlo, int khi,
    v8f& acc0, v8f& acc1)
{
    FragU a, b0, b1;
    // A: lane<16 holds K 0..15, lane>=16 holds K 16..31 (contiguous, matches B)
    const unsigned short* ap = A + (m0 + nlo) * lda + kc + khi * 16;
    a.u[0] = *(const uint4*)(ap);
    a.u[1] = *(const uint4*)(ap + 8);
    // B: lane holds column n = nlo, 16 contiguous K at kc + khi*16
    const unsigned short* bp0 = B + (col0 + nlo) * ldb + kc + khi * 16;
    b0.u[0] = ((const uint4*)bp0)[0];
    b0.u[1] = ((const uint4*)bp0)[1];
    const unsigned short* bp1 = B + (col1 + nlo) * ldb + kc + khi * 16;
    b1.u[0] = ((const uint4*)bp1)[0];
    b1.u[1] = ((const uint4*)bp1)[1];

    acc0 = __builtin_amdgcn_wmma_f32_16x16x32_bf16(false, a.v, false, b0.v,
                                                   (short)0, acc0, false, false);
    acc1 = __builtin_amdgcn_wmma_f32_16x16x32_bf16(false, a.v, false, b1.v,
                                                   (short)0, acc1, false, false);
}

// ------------------------------ prep kernels -------------------------------

__global__ void cast_bf16_kernel(const float* __restrict__ s,
                                 unsigned short* __restrict__ d, int n) {
    int i = blockIdx.x * blockDim.x + threadIdx.x;
    int stride = gridDim.x * blockDim.x;
    for (; i < n; i += stride) d[i] = f2bf(s[i]);
}

__global__ void embed_bf16_kernel(const int* __restrict__ tok,
                                  const float* __restrict__ tab,
                                  unsigned short* __restrict__ d, int n) {
    int i = blockIdx.x * blockDim.x + threadIdx.x;
    int stride = gridDim.x * blockDim.x;
    for (; i < n; i += stride) {
        int tb = i >> 6;            // (t*B + b)
        int c  = i & 63;            // ninp index
        d[i] = f2bf(tab[tok[tb] * NINP + c]);
    }
}

__global__ void init_state_kernel(unsigned short* __restrict__ h0,
                                  unsigned* __restrict__ cnt) {
    int i = blockIdx.x * blockDim.x + threadIdx.x;
    int stride = gridDim.x * blockDim.x;
    for (; i < BATCH * NHID; i += stride) h0[i] = 0;   // h buffer 0 = zeros
    if (blockIdx.x == 0 && threadIdx.x == 0) *cnt = 0u;
}

// --------------------------- persistent LSTM kernel ------------------------

static __device__ __forceinline__ void grid_sync(unsigned* cnt, unsigned target) {
    __syncthreads();
    if (threadIdx.x == 0) {
        __threadfence();  // release h writes to device scope
        __hip_atomic_fetch_add(cnt, 1u, __ATOMIC_RELEASE, __HIP_MEMORY_SCOPE_AGENT);
        while (__hip_atomic_load(cnt, __ATOMIC_ACQUIRE, __HIP_MEMORY_SCOPE_AGENT) < target) {
            __builtin_amdgcn_s_sleep(1);
        }
        __threadfence();
    }
    __syncthreads();
}

__global__ __launch_bounds__(256) void lstm_persistent_kernel(
    const unsigned short* __restrict__ whh,    // [16384][4096] bf16
    const unsigned short* __restrict__ wih,    // [16384][64]   bf16
    const unsigned short* __restrict__ embt,   // [256][64][64] bf16
    const float* __restrict__ b_ih,            // [16384]
    const float* __restrict__ b_hh,            // [16384]
    const int*   __restrict__ seq_len,         // [64]
    unsigned short* __restrict__ hbuf,         // 2 x [64][4096] bf16 (double buffer)
    unsigned* __restrict__ counter,
    float* __restrict__ out)                   // [64][4096] f32 (feat = c)
{
    __shared__ float gl[4][BATCH][17];         // gates i,f,g,o  [gate][b][j-in-tile]

    const int tid  = threadIdx.x;
    const int lane = tid & 31;
    const int w    = tid >> 5;                 // wave 0..7
    const int mt   = w & 3;                    // M-tile (batch rows)
    const int m0   = mt * 16;
    const int g0   = (w >> 2) * 2;             // gates {g0, g0+1}
    const int j0   = blockIdx.x * 16;          // hidden-unit tile base
    const int nlo  = lane & 15;
    const int khi  = lane >> 4;

    const int col0 = g0 * NHID + j0;
    const int col1 = (g0 + 1) * NHID + j0;
    const float bias0 = b_ih[col0 + nlo] + b_hh[col0 + nlo];
    const float bias1 = b_ih[col1 + nlo] + b_hh[col1 + nlo];

    // per-thread elementwise state: 4 (b, j) pairs
    float c_reg[4], feat_reg[4];
    int   sl[4], bb[4], jj[4];
#pragma unroll
    for (int k = 0; k < 4; ++k) {
        int p = tid * 4 + k;
        bb[k] = p >> 4;
        jj[k] = p & 15;
        sl[k] = seq_len[bb[k]];
        c_reg[k] = 0.0f;
        feat_reg[k] = 0.0f;
    }

    const unsigned NWG = gridDim.x;

    for (int t = 0; t < T_STEPS; ++t) {
        const unsigned short* hread  = hbuf + (unsigned)(t & 1) * (BATCH * NHID);
        unsigned short*       hwrite = hbuf + (unsigned)((t + 1) & 1) * (BATCH * NHID);

        v8f acc0, acc1;
#pragma unroll
        for (int r = 0; r < 8; ++r) { acc0[r] = bias0; acc1[r] = bias1; }

        // fused input projection: A = emb[t] [64 x 64], B = w_ih (ldb = 64)
        const unsigned short* Ae = embt + (unsigned)t * (BATCH * NINP);
#pragma unroll
        for (int kc = 0; kc < NINP; kc += 32)
            kstep(Ae, NINP, wih, NINP, col0, col1, kc, m0, nlo, khi, acc0, acc1);

        // recurrent GEMM: A = h [64 x 4096], B = w_hh (ldb = 4096)
        for (int kc = 0; kc < NHID; kc += 32)
            kstep(hread, NHID, whh, NHID, col0, col1, kc, m0, nlo, khi, acc0, acc1);

        // C/D layout: element (vgpr r, lane) -> m = r + 8*khi (+ m0), n = nlo
#pragma unroll
        for (int r = 0; r < 8; ++r) {
            int m = m0 + r + 8 * khi;
            gl[g0][m][nlo]     = acc0[r];
            gl[g0 + 1][m][nlo] = acc1[r];
        }
        __syncthreads();

#pragma unroll
        for (int k = 0; k < 4; ++k) {
            int b = bb[k], j = jj[k];
            float ig = gl[0][b][j];
            float fg = gl[1][b][j];
            float gg = gl[2][b][j];
            float og = gl[3][b][j];
            float c = sigm(fg) * c_reg[k] + sigm(ig) * tanhf(gg);
            c_reg[k] = c;
            float h = sigm(og) * tanhf(c);
            bool valid = (t < sl[k]) || (t == 0);
            if (valid) feat_reg[k] = c;
            hwrite[b * NHID + j0 + j] = f2bf(h);
        }

        grid_sync(counter, NWG * (unsigned)(t + 1));
    }

#pragma unroll
    for (int k = 0; k < 4; ++k)
        out[bb[k] * NHID + j0 + jj[k]] = feat_reg[k];
}

// --------------------------------- launch ----------------------------------

extern "C" void kernel_launch(void* const* d_in, const int* in_sizes, int n_in,
                              void* d_out, int out_size, void* d_ws, size_t ws_size,
                              hipStream_t stream) {
    const int*   tokens    = (const int*)d_in[0];
    const int*   seq_len   = (const int*)d_in[1];
    const float* emb_table = (const float*)d_in[2];
    const float* w_ih      = (const float*)d_in[3];
    const float* w_hh      = (const float*)d_in[4];
    const float* b_ih      = (const float*)d_in[5];
    const float* b_hh      = (const float*)d_in[6];

    char* ws = (char*)d_ws;
    unsigned short* whh_bf = (unsigned short*)(ws + WHH_OFF);
    unsigned short* wih_bf = (unsigned short*)(ws + WIH_OFF);
    unsigned short* emb_bf = (unsigned short*)(ws + EMB_OFF);
    unsigned short* hbuf   = (unsigned short*)(ws + HBUF_OFF);
    unsigned*       cnt    = (unsigned*)(ws + CNT_OFF);

    // bf16 conversions (one HBM pass; w_hh bf16 = 128 MB -> L2 resident afterwards)
    cast_bf16_kernel<<<2048, 256, 0, stream>>>(w_hh, whh_bf, GATES * NHID);
    cast_bf16_kernel<<<512, 256, 0, stream>>>(w_ih, wih_bf, GATES * NINP);
    embed_bf16_kernel<<<512, 256, 0, stream>>>(tokens, emb_table, emb_bf,
                                               T_STEPS * BATCH * NINP);
    init_state_kernel<<<64, 256, 0, stream>>>(hbuf, cnt);

    // persistent recurrence: 256 WGs (16 hidden units each) x 256 threads
    lstm_persistent_kernel<<<NHID / 16, 256, 0, stream>>>(
        whh_bf, wih_bf, emb_bf, b_ih, b_hh, seq_len, hbuf, cnt, (float*)d_out);
}
